// RetinaFaceDetector_29618094473286
// MI455X (gfx1250) — compile-verified, matrix operand order checked by reading, not run on hardware
//
#include <hip/hip_runtime.h>
#include <stdint.h>

typedef _Float16 v16h __attribute__((ext_vector_type(16)));
typedef float    v8f  __attribute__((ext_vector_type(8)));
typedef int      rf_i4 __attribute__((vector_size(16)));   // matches builtin param type
typedef __attribute__((address_space(1))) rf_i4* rf_g4p;   // global (device) AS
typedef __attribute__((address_space(3))) rf_i4* rf_l4p;   // LDS AS

#define RF_VAR0   0.1f
#define RF_VAR1   0.2f
#define RF_CONF_T 0.5f
#define RF_IOU_T  0.3f
#define RF_SCALE  512.0f
#define RF_CH     4096          // LDS bitonic chunk (32KB of u64 keys)

#if __has_builtin(__builtin_amdgcn_global_load_async_to_lds_b128)
#define RF_ASYNC_LDS 1
#else
#define RF_ASYNC_LDS 0
#endif

static __device__ __forceinline__ void rf_wait_async0() {
#if __has_builtin(__builtin_amdgcn_s_wait_asynccnt)
    __builtin_amdgcn_s_wait_asynccnt(0);
#else
    asm volatile("s_wait_asynccnt 0x0" ::: "memory");
#endif
}

// ---------------------------------------------------------------------------
// init: zero output, keep flags, valid counters
// ---------------------------------------------------------------------------
__global__ void rf_init(float* out, int outN, int* keep, int keepN, int* vcnt, int B) {
    int t = blockIdx.x * blockDim.x + threadIdx.x;
    if (t < outN)  out[t]  = 0.0f;
    if (t < keepN) keep[t] = 0;
    if (t < B)     vcnt[t] = 0;
}

// ---------------------------------------------------------------------------
// decode: boxes (xyxy, normalized), sigmoid scores, sortable keys
// key = (~orderable(score)) << 32 | idx  -> ascending u64 sort == stable
// argsort(-score) with index tiebreak. Pad slots get key = ~0.
// ---------------------------------------------------------------------------
__global__ void rf_decode(const float* __restrict__ loc, const float* __restrict__ conf,
                          const float* __restrict__ priors, float* __restrict__ boxes,
                          float* __restrict__ scores, unsigned long long* __restrict__ keys,
                          int* __restrict__ vcnt, int N, int SORT, int B) {
    int t = blockIdx.x * blockDim.x + threadIdx.x;
    int b = t / SORT, i = t - b * SORT;
    if (b >= B) return;
    unsigned long long key;
    if (i < N) {
        float4 p = *(const float4*)(priors + (size_t)i * 4);
        float4 l = *(const float4*)(loc + ((size_t)b * N + i) * 4);
        float cx = p.x + l.x * RF_VAR0 * p.z;
        float cy = p.y + l.y * RF_VAR0 * p.w;
        float w  = p.z * expf(l.z * RF_VAR1);
        float h  = p.w * expf(l.w * RF_VAR1);
        float x1 = cx - 0.5f * w, y1 = cy - 0.5f * h;
        float4 bx = make_float4(x1, y1, x1 + w, y1 + h);
        *(float4*)(boxes + ((size_t)b * N + i) * 4) = bx;
        float c = conf[((size_t)b * N + i) * 2 + 1];
        float s = 1.0f / (1.0f + expf(-c));
        scores[(size_t)b * N + i] = s;
        unsigned u = __float_as_uint(s);
        u = (u & 0x80000000u) ? ~u : (u | 0x80000000u);   // orderable ascending
        key = ((unsigned long long)(~u) << 32) | (unsigned)i;
        if (s >= RF_CONF_T) atomicAdd(vcnt + b, 1);
    } else {
        key = ~0ULL;                                      // pad -> sorts last
    }
    keys[(size_t)b * SORT + i] = key;
}

// ---------------------------------------------------------------------------
// global bitonic compare-exchange pass (only for j >= CH)
// ---------------------------------------------------------------------------
__global__ void rf_bitonic(unsigned long long* keys, int k, int j, int SORT, int B) {
    int t = blockIdx.x * blockDim.x + threadIdx.x;
    int b = t / SORT, i = t - b * SORT;
    if (b >= B) return;
    int ixj = i ^ j;
    if (ixj <= i) return;
    unsigned long long* base = keys + (size_t)b * SORT;
    unsigned long long a = base[i], c = base[ixj];
    bool up = (i & k) == 0;
    if (up ? (a > c) : (a < c)) { base[i] = c; base[ixj] = a; }
}

// ---------------------------------------------------------------------------
// fused LDS bitonic: one workgroup sorts a CH-key chunk in LDS, running all
// stages k=kStart..kEnd with j<=CH/2 (directions taken from the GLOBAL index,
// so this is bit-identical to the global network). Fuses 78 of the 105
// passes (SORT=16384) into one launch; tails of k=8192/16384 fuse 12 each.
// ---------------------------------------------------------------------------
__global__ void __launch_bounds__(1024)
rf_bitonic_lds(unsigned long long* keys, int SORT, int CH, int kStart, int kEnd) {
    __shared__ unsigned long long sk[RF_CH];
    int perImg = SORT / CH;
    int chunk  = blockIdx.x;
    int c = chunk % perImg;
    int b = chunk / perImg;
    unsigned long long* base = keys + (size_t)b * SORT + (size_t)c * CH;
    int tid = threadIdx.x;
    for (int e = tid; e < CH; e += (int)blockDim.x) sk[e] = base[e];
    __syncthreads();
    int gbase = c * CH;                         // within-image index of element 0
    for (int k = kStart; k <= kEnd; k <<= 1) {
        int j0 = ((k >> 1) < CH) ? (k >> 1) : (CH >> 1);
        for (int j = j0; j > 0; j >>= 1) {
            for (int t = tid; t < (CH >> 1); t += (int)blockDim.x) {
                int e = ((t & ~(j - 1)) << 1) | (t & (j - 1));
                int p = e | j;
                bool up = (((unsigned)(gbase + e)) & (unsigned)k) == 0;
                unsigned long long a = sk[e], d = sk[p];
                if (up ? (a > d) : (a < d)) { sk[e] = d; sk[p] = a; }
            }
            __syncthreads();
        }
    }
    for (int e = tid; e < CH; e += (int)blockDim.x) base[e] = sk[e];
}

// ---------------------------------------------------------------------------
// gather top-M sorted boxes + areas
// ---------------------------------------------------------------------------
__global__ void rf_gather(const unsigned long long* __restrict__ keys,
                          const float* __restrict__ boxes,
                          float* __restrict__ sboxes, float* __restrict__ sarea,
                          int M, int N, int SORT, int B) {
    int t = blockIdx.x * blockDim.x + threadIdx.x;
    if (t >= B * M) return;
    int b = t / M, i = t - b * M;
    unsigned long long key = keys[(size_t)b * SORT + i];
    unsigned idx = (unsigned)key;
    float4 bx = make_float4(0.f, 0.f, 0.f, 0.f);
    if (idx < (unsigned)N) bx = *(const float4*)(boxes + ((size_t)b * N + idx) * 4);
    *(float4*)(sboxes + ((size_t)b * M + i) * 4) = bx;
    sarea[(size_t)b * M + i] = fmaxf(bx.z - bx.x, 0.f) * fmaxf(bx.w - bx.y, 0.f);
}

// ---------------------------------------------------------------------------
// IoU bitmask, one 16x16 tile per wave32.
// Tile boxes are DMA'd into LDS with GLOBAL_LOAD_ASYNC_TO_LDS_B128 (one 16B
// box per lane, tracked by ASYNCcnt); the wave overlaps the copy with the
// union outer-sum on the matrix pipe:
//   A[m] = [hiA_m, loA_m, 1, 1, 0...]   (hi/lo f16 split -> f32-exact sum)
//   B[:,n] = [1, 1, hiB_n, loB_n, 0...]
//   D[m][n] = area_m + area_n via V_WMMA_F32_16X16X32_F16 (f32 accumulate)
// then waits s_wait_asynccnt 0 and finishes inter/union with VALU min/max.
// Each wave reads only its own staged tiles -> no block barrier needed.
// C/D layout: lane L -> col n=L%16, VGPR v -> row m=v+8*(L/16), so one
// __ballot per v yields two 16-bit mask rows; stored as disjoint u16 words.
// ---------------------------------------------------------------------------
__global__ void __launch_bounds__(256)
rf_mask(const float* __restrict__ sboxes, const float* __restrict__ sarea,
        unsigned short* __restrict__ mask, int M, int T, long long nTiles) {
    __shared__ float4 shRow[8][16];
    __shared__ float4 shCol[8][16];
    int lane = threadIdx.x & 31;
    int wave = threadIdx.x >> 5;
    long long tile = (long long)blockIdx.x * 8 + wave;
    if (tile >= nTiles) tile = nTiles - 1;       // duplicate work, keeps EXEC full
    int tj = (int)(tile % T);
    long long rr = tile / T;
    int ti = (int)(rr % T);
    int b  = (int)(rr / T);
    int rowBase = ti * 16, colBase = tj * 16;
    const float* SB = sboxes + (size_t)b * M * 4;
    const float* SA = sarea  + (size_t)b * M;
    __builtin_prefetch(SB + (size_t)rowBase * 4, 0, 0);
    __builtin_prefetch(SB + (size_t)colBase * 4, 0, 0);

    int l16 = lane & 15;
    int hi  = lane >> 4;

#if RF_ASYNC_LDS
    {   // lanes 0-15 DMA row boxes, lanes 16-31 DMA col boxes, LDS-direct
        const float* src = (hi == 0) ? (SB + (size_t)(rowBase + l16) * 4)
                                     : (SB + (size_t)(colBase + l16) * 4);
        float4* dst = (hi == 0) ? &shRow[wave][l16] : &shCol[wave][l16];
        __builtin_amdgcn_global_load_async_to_lds_b128((rf_g4p)src, (rf_l4p)dst, 0, 0);
    }
    float aR = 0.f, aC = 0.f;
    if (hi == 0) { aR = SA[rowBase + l16]; aC = SA[colBase + l16]; }
#else
    float aR = 0.f, aC = 0.f;
    if (hi == 0) {
        shRow[wave][l16] = *(const float4*)(SB + (size_t)(rowBase + l16) * 4);
        aR = SA[rowBase + l16];
        aC = SA[colBase + l16];
    } else {
        shCol[wave][l16] = *(const float4*)(SB + (size_t)(colBase + l16) * 4);
    }
#endif

    v16h A, Bv;
#pragma unroll
    for (int e = 0; e < 16; ++e) { A[e] = (_Float16)0.0f; Bv[e] = (_Float16)0.0f; }
    if (hi == 0) {
        _Float16 hA = (_Float16)aR; _Float16 lA = (_Float16)(aR - (float)hA);
        _Float16 hB = (_Float16)aC; _Float16 lB = (_Float16)(aC - (float)hB);
        A[0]  = hA; A[1]  = lA; A[2]  = (_Float16)1.0f; A[3]  = (_Float16)1.0f;
        Bv[0] = (_Float16)1.0f; Bv[1] = (_Float16)1.0f; Bv[2] = hB; Bv[3] = lB;
    }
    v8f C = {};
    C = __builtin_amdgcn_wmma_f32_16x16x32_f16(false, A, false, Bv,
                                               (short)0, C, false, false);

#if RF_ASYNC_LDS
    rf_wait_async0();                            // tiles landed in LDS
#else
    asm volatile("s_wait_dscnt 0x0" ::: "memory");
#endif

    float4 cb = shCol[wave][l16];
    unsigned bal[8];
#pragma unroll
    for (int v = 0; v < 8; ++v) {
        int m = v + 8 * hi;
        float4 rb = shRow[wave][m];
        float ix1 = fmaxf(rb.x, cb.x), iy1 = fmaxf(rb.y, cb.y);
        float ix2 = fminf(rb.z, cb.z), iy2 = fminf(rb.w, cb.w);
        float inter = fmaxf(ix2 - ix1, 0.f) * fmaxf(iy2 - iy1, 0.f);
        float uni = C[v] - inter;                 // area_i + area_j - inter
        bool pred = inter > RF_IOU_T * uni;       // iou > T (div-free)
        bal[v] = (unsigned)__ballot(pred);        // lo16: row v, hi16: row v+8
    }
    if (hi == 0) {
        unsigned wbits = (l16 < 8) ? (bal[l16] & 0xFFFFu) : (bal[l16 - 8] >> 16);
        mask[((size_t)b * M + (rowBase + l16)) * (size_t)(M / 16) + tj] =
            (unsigned short)wbits;
    }
}

// ---------------------------------------------------------------------------
// greedy scan: one wave32 per image, remv bitmask in LDS. Only valid boxes
// (sorted prefix of length vcnt) can be kept; kept rows OR their mask row.
// ---------------------------------------------------------------------------
__global__ void rf_scan(const unsigned short* __restrict__ mask,
                        const int* __restrict__ vcnt, int* __restrict__ keep, int M) {
    int b = blockIdx.x;
    int lane = threadIdx.x;
    __shared__ unsigned remv[512];
    int Mw = M >> 5;
    for (int w = lane; w < Mw; w += 32) remv[w] = 0u;
    __syncthreads();
    int vc = vcnt[b]; if (vc > M) vc = M;
    const unsigned* mrow = (const unsigned*)(mask + (size_t)b * M * (size_t)(M / 16));
    for (int i = 0; i < vc; ++i) {
        unsigned sup = (remv[i >> 5] >> (i & 31)) & 1u;   // uniform across wave
        if (!sup) {
            if (lane == 0) keep[(size_t)b * M + i] = 1;
            const unsigned* row = mrow + (size_t)i * Mw;
            for (int w = lane; w < Mw; w += 32) remv[w] |= row[w];
        }
        __syncthreads();
    }
}

// ---------------------------------------------------------------------------
// scatter kept detections to original slots: [box*512, score]
// ---------------------------------------------------------------------------
__global__ void rf_scatter(const int* __restrict__ keep,
                           const unsigned long long* __restrict__ keys,
                           const float* __restrict__ boxes, const float* __restrict__ scores,
                           float* __restrict__ out, int M, int N, int SORT, int B) {
    int t = blockIdx.x * blockDim.x + threadIdx.x;
    if (t >= B * M) return;
    if (!keep[t]) return;
    int b = t / M, i = t - b * M;
    unsigned long long key = keys[(size_t)b * SORT + i];
    unsigned idx = (unsigned)key;
    if (idx >= (unsigned)N) return;
    float4 bx = *(const float4*)(boxes + ((size_t)b * N + idx) * 4);
    float s = scores[(size_t)b * N + idx];
    float* o = out + ((size_t)b * N + idx) * 5;
    o[0] = bx.x * RF_SCALE; o[1] = bx.y * RF_SCALE;
    o[2] = bx.z * RF_SCALE; o[3] = bx.w * RF_SCALE;
    o[4] = s;
}

// ---------------------------------------------------------------------------
// host side
// ---------------------------------------------------------------------------
struct RfLayout {
    size_t boxes, scores, keys, sboxes, sarea, keep, vcnt, mask, total;
};

static RfLayout rf_layout(int B, int N, int SORT, int M) {
    RfLayout L;
    size_t off = 0;
    auto al = [&](size_t bytes) {
        off = (off + 255) & ~(size_t)255;
        size_t o = off; off += bytes; return o;
    };
    L.boxes  = al((size_t)B * N * 4 * sizeof(float));
    L.scores = al((size_t)B * N * sizeof(float));
    L.keys   = al((size_t)B * SORT * sizeof(unsigned long long));
    L.sboxes = al((size_t)B * M * 4 * sizeof(float));
    L.sarea  = al((size_t)B * M * sizeof(float));
    L.keep   = al((size_t)B * M * sizeof(int));
    L.vcnt   = al((size_t)B * sizeof(int));
    L.mask   = al((size_t)B * M * (size_t)(M / 16) * sizeof(unsigned short));
    L.total  = off;
    return L;
}

extern "C" void kernel_launch(void* const* d_in, const int* in_sizes, int n_in,
                              void* d_out, int out_size, void* d_ws, size_t ws_size,
                              hipStream_t stream) {
    const float* loc    = (const float*)d_in[0];
    const float* conf   = (const float*)d_in[1];
    const float* priors = (const float*)d_in[2];
    int N = in_sizes[2] / 4;                         // priors [N,4]
    if (N <= 0) return;
    int B = in_sizes[0] / (N * 4);                   // loc [B,N,4]
    if (B <= 0) return;
    int SORT = 1; while (SORT < N) SORT <<= 1;       // 16384 for N=10752

    // NMS candidate cap: valid-count is Binomial(N, ~1/2) -> 6144 is >+14 sigma.
    int M = 6144; if (M > SORT) M = SORT;
    while (M > 512 && rf_layout(B, N, SORT, M).total > ws_size) M -= 512;
    RfLayout L = rf_layout(B, N, SORT, M);

    char* w = (char*)d_ws;
    float*              boxes  = (float*)(w + L.boxes);
    float*              scores = (float*)(w + L.scores);
    unsigned long long* keys   = (unsigned long long*)(w + L.keys);
    float*              sboxes = (float*)(w + L.sboxes);
    float*              sarea  = (float*)(w + L.sarea);
    int*                keep   = (int*)(w + L.keep);
    int*                vcnt   = (int*)(w + L.vcnt);
    unsigned short*     maskp  = (unsigned short*)(w + L.mask);
    float*              out    = (float*)d_out;

    int initN = out_size;
    if (B * M > initN) initN = B * M;
    if (B > initN) initN = B;
    rf_init<<<(initN + 255) / 256, 256, 0, stream>>>(out, out_size, keep, B * M, vcnt, B);

    rf_decode<<<(B * SORT + 255) / 256, 256, 0, stream>>>(
        loc, conf, priors, boxes, scores, keys, vcnt, N, SORT, B);

    // ---- sort: fused LDS chunks + sparse global passes (6 launches total) ----
    int CH = RF_CH; if (CH > SORT) CH = SORT;
    int nChunkBlk = (B * SORT) / CH;
    rf_bitonic_lds<<<nChunkBlk, 1024, 0, stream>>>(keys, SORT, CH, 2, CH);
    for (int k = CH << 1; k <= SORT; k <<= 1) {
        for (int j = k >> 1; j >= CH; j >>= 1)
            rf_bitonic<<<(B * SORT + 255) / 256, 256, 0, stream>>>(keys, k, j, SORT, B);
        rf_bitonic_lds<<<nChunkBlk, 1024, 0, stream>>>(keys, SORT, CH, k, k);
    }

    rf_gather<<<(B * M + 255) / 256, 256, 0, stream>>>(keys, boxes, sboxes, sarea,
                                                       M, N, SORT, B);

    int T = M / 16;
    long long nTiles = (long long)B * T * T;
    long long nBlk = (nTiles + 7) / 8;               // 8 waves (tiles) per block
    rf_mask<<<(unsigned)nBlk, 256, 0, stream>>>(sboxes, sarea, maskp, M, T, nTiles);

    rf_scan<<<B, 32, 0, stream>>>(maskp, vcnt, keep, M);

    rf_scatter<<<(B * M + 255) / 256, 256, 0, stream>>>(keep, keys, boxes, scores,
                                                        out, M, N, SORT, B);
}
